// CauseMutiHeadAttention_73512660238764
// MI455X (gfx1250) — matrix operand order repaired and checked by reading, not compiled, not run on hardware
//
#include <hip/hip_runtime.h>
#include <hip/hip_bf16.h>
#include <math.h>

// ---------------------------------------------------------------------------
// Causal MHA with RoPE for MI455X (gfx1250, wave32, WMMA bf16 -> f32 acc)
//   B=2, S=2048, D=1024, H=16, DK=64
// GEMMs via v_wmma_f32_16x16x32_bf16; flash-attention K/V staged into LDS
// with the Tensor Data Mover (tensor_load_to_lds, TENSORcnt), double-buffered.
// ---------------------------------------------------------------------------

#define DEVINL __device__ __forceinline__

typedef __attribute__((ext_vector_type(16))) __bf16 bf16x16;
typedef __attribute__((ext_vector_type(8)))  float  f32x8;
typedef __attribute__((ext_vector_type(4))) unsigned tdm_g0_t;
typedef __attribute__((ext_vector_type(8))) int      tdm_g1_t;
typedef __attribute__((ext_vector_type(4))) int      tdm_g2_t;
typedef __attribute__((ext_vector_type(8))) int      tdm_g4_t;

union BFVec { bf16x16 v; uint4 q[2]; unsigned short s[16]; };

static constexpr int B_ = 2, S_ = 2048, D_ = 1024, H_ = 16, DK_ = 64;
static constexpr int M_ = B_ * S_;                // 4096 total rows
static constexpr float SCALE_ = 0.125f;           // 1/sqrt(64)
static constexpr float LOG_THETA_ = 9.210340371976184f; // ln(10000)

DEVINL unsigned short f32_to_bf16(float f) {
  unsigned u = __builtin_bit_cast(unsigned, f);
  u += 0x7FFFu + ((u >> 16) & 1u);                // round-to-nearest-even
  return (unsigned short)(u >> 16);
}
DEVINL unsigned pack_bf16x2(float a, float b) {
  return (unsigned)f32_to_bf16(a) | ((unsigned)f32_to_bf16(b) << 16);
}
DEVINL f32x8 wmma_bf16(const BFVec& a, const BFVec& b, f32x8 c) {
  return __builtin_amdgcn_wmma_f32_16x16x32_bf16(false, a.v, false, b.v,
                                                 (short)0, c, false, false);
}

// ---------------------------------------------------------------------------
// TDM: async DMA of a 2D bf16 tile [tile_rows x row_elems] (row-major,
// stride row_elems) from global memory into LDS at byte offset lds_off.
// D# built per CDNA5 ISA (group0 128b, group1 256b).
// This toolchain exposes the 6-arg builtin:
//   (uint32x4 g0, int32x8 g1, int32x4 g2, int32x4 g3, int32x8 pad, i32 cpol)
// ---------------------------------------------------------------------------
DEVINL void tdm_load_2d(unsigned lds_off, const void* gptr,
                        unsigned tile_rows, unsigned row_elems,
                        unsigned rows_total) {
  unsigned long long ga = (unsigned long long)(size_t)gptr;
  tdm_g0_t g0;
  g0[0] = 1u;                                   // count=1 (valid), user mode
  g0[1] = lds_off;                              // lds_addr (bytes)
  g0[2] = (unsigned)ga;                         // global_addr[31:0]
  g0[3] = (unsigned)((ga >> 32) & 0x01FFFFFFu)  // global_addr[56:32]
        | (2u << 30);                           // type=2 ("image")
  tdm_g1_t g1;
  g1[0] = (int)(1u << 16);                      // wg_mask=0, data_size=1 (2B)
  g1[1] = (int)(row_elems << 16);               // tensor_dim0[15:0] @ bits63:48
  g1[2] = (int)((rows_total & 0xFFFFu) << 16);  // dim0 hi=0 | tensor_dim1 lo
  g1[3] = (int)(row_elems << 16);               // dim1 hi=0 | tile_dim0
  g1[4] = (int)(tile_rows & 0xFFFFu);           // tile_dim1 | tile_dim2=0
  g1[5] = (int)row_elems;                       // tensor_dim0_stride[31:0]
  g1[6] = 0;                                    // stride0 hi | stride1 lo
  g1[7] = 0;                                    // stride1 hi
  tdm_g2_t z4 = {0, 0, 0, 0};                   // groups 2/3 unused (2D)
  tdm_g4_t z8 = {0, 0, 0, 0, 0, 0, 0, 0};
  __builtin_amdgcn_tensor_load_to_lds(g0, g1, z4, z4, z8, 0);
}

// ---------------------------------------------------------------------------
// f32 -> bf16 bulk convert (vectorized x4)
// ---------------------------------------------------------------------------
__global__ void __launch_bounds__(256)
cvt_f32_bf16(const float* __restrict__ src, unsigned short* __restrict__ dst, int n) {
  int i = (blockIdx.x * 256 + threadIdx.x) * 4;
  if (i + 3 < n) {
    float4 f = *(const float4*)(src + i);
    uint2 o;
    o.x = pack_bf16x2(f.x, f.y);
    o.y = pack_bf16x2(f.z, f.w);
    *(uint2*)(dst + i) = o;
  }
}

// ---------------------------------------------------------------------------
// One 16x16 output tile of [M_,D_] x [D_,D_] via v_wmma_f32_16x16x32_bf16.
// ---------------------------------------------------------------------------
DEVINL f32x8 gemm16x16(const unsigned short* __restrict__ A,
                       const unsigned short* __restrict__ Bm,
                       int m0, int n0, int lane) {
  f32x8 acc = {0.f,0.f,0.f,0.f,0.f,0.f,0.f,0.f};
  const int arow = m0 + (lane & 15);
  const int ak   = (lane >> 4) * 8;
  for (int k0 = 0; k0 < D_; k0 += 32) {
    BFVec a, b;
    const unsigned short* ap = A + (size_t)arow * D_ + k0 + ak;
    a.q[0] = *(const uint4*)(ap);
    a.q[1] = *(const uint4*)(ap + 16);
    const unsigned short* bp = Bm + (size_t)(k0 + ak) * D_ + n0 + (lane & 15);
    __builtin_prefetch(bp + 64 * D_, 0, 1);     // global_prefetch k-chunk two steps ahead
    for (int s = 0; s < 8; ++s) {
      b.s[s] = bp[(size_t)s * D_];
      b.s[8 + s] = bp[(size_t)(16 + s) * D_];
    }
    acc = wmma_bf16(a, b, acc);
  }
  return acc;
}

// ---------------------------------------------------------------------------
// Fused QKV projection + bias + RoPE (Q,K) epilogue. Output bf16 [b,h,s,dk].
// ---------------------------------------------------------------------------
__global__ void __launch_bounds__(256)
qkv_rope_gemm(const unsigned short* __restrict__ xbf,
              const unsigned short* __restrict__ wq,
              const unsigned short* __restrict__ wk,
              const unsigned short* __restrict__ wv,
              const float* __restrict__ bq, const float* __restrict__ bk,
              const float* __restrict__ bv,
              const int* __restrict__ tokpos,
              unsigned short* __restrict__ Qo,
              unsigned short* __restrict__ Ko,
              unsigned short* __restrict__ Vo) {
  const int lane = threadIdx.x & 31;
  const int wave = threadIdx.x >> 5;
  const int gt   = blockIdx.x * 8 + wave;
  const int TN = D_ / 16, TM = M_ / 16;
  const int per = TM * TN;
  const int which = gt / per;
  const int rem   = gt % per;
  const int m0 = (rem / TN) * 16;
  const int n0 = (rem % TN) * 16;

  const unsigned short* W   = (which == 0) ? wq : (which == 1) ? wk : wv;
  const float*          bia = (which == 0) ? bq : (which == 1) ? bk : bv;
  unsigned short*       Out = (which == 0) ? Qo : (which == 1) ? Ko : Vo;

  f32x8 acc = gemm16x16(xbf, W, m0, n0, lane);

  const int n  = n0 + (lane & 15);
  const float bn = bia[n];
  const int h  = n >> 6;
  const int dk = n & 63;
  const float freq = __expf(-(float)(dk & ~1) * (LOG_THETA_ / (float)DK_));

#pragma unroll
  for (int r = 0; r < 8; ++r) {
    const int m  = m0 + r + 8 * (lane >> 4);
    const int bb = m / S_;
    const int s  = m - bb * S_;
    float v = acc[r] + bn;
    if (which < 2) {
      const float ang = (float)tokpos[m] * freq;
      const float c  = __cosf(ang);
      const float sn = __sinf(ang);
      const float partner = __shfl_xor(v, 1, 32);
      v = ((dk & 1) == 0) ? (v * c - partner * sn)
                          : (partner * sn + v * c);
    }
    Out[(((size_t)(bb * H_ + h)) * S_ + s) * DK_ + dk] = f32_to_bf16(v);
  }
}

// ---------------------------------------------------------------------------
// Flash attention. Block = 8 waves = 128 query rows of one (b,h).
// K/V blocks (32 keys x 64 dk, bf16) are DMA'd into LDS by the TDM
// (issued by wave 0, double-buffered, TENSORcnt + workgroup barrier),
// shared by all 8 waves. Scores computed transposed (S^T = K @ Q^T) so all
// WMMA operands come from contiguous LDS rows; Q^T staged once per wave.
// ---------------------------------------------------------------------------
__global__ void __launch_bounds__(256)
flash_attn(const unsigned short* __restrict__ Q,
           const unsigned short* __restrict__ K,
           const unsigned short* __restrict__ V,
           unsigned short* __restrict__ O) {
  // LDS arena (byte offsets are also used as TDM D#.lds_addr):
  //   [0,16K)   qT: per-wave Q^T [dk64][q16] bf16      (8 x 2048B)
  //   [16K,24K) pS: per-wave P   [q16][key32] bf16     (8 x 1024B)
  //   [24K,32K) kT: K tile double buffer [2][32][64]   (2 x 4096B)
  //   [32K,40K) vT: V tile double buffer [2][32][64]   (2 x 4096B)
  __shared__ __align__(16) unsigned char smem[40960];
  constexpr unsigned QT_OFF = 0, PS_OFF = 16384, KT_OFF = 24576, VT_OFF = 32768;

  const int lane = threadIdx.x & 31;
  const int wave = threadIdx.x >> 5;
  const int b = blockIdx.z, h = blockIdx.y;
  const int qr = blockIdx.x * 128 + wave * 16;

  const unsigned short* Qb = Q + (size_t)(b * H_ + h) * S_ * DK_;
  const unsigned short* Kb = K + (size_t)(b * H_ + h) * S_ * DK_;
  const unsigned short* Vb = V + (size_t)(b * H_ + h) * S_ * DK_;

  unsigned short* qTw = (unsigned short*)(smem + QT_OFF + wave * 2048);
  unsigned short* pSw = (unsigned short*)(smem + PS_OFF + wave * 1024);

  // ---- kick off TDM for key block 0 (K and V) ----
  if (wave == 0) {
    tdm_load_2d(KT_OFF, Kb, 32, DK_, S_);
    tdm_load_2d(VT_OFF, Vb, 32, DK_, S_);
  }

  // ---- stage Q^T into per-wave LDS (one-time transpose) ----
  {
    const int qrow = lane & 15;
    const int koff = (lane >> 4) * 32;  // lanes 0-15: dk 0..31; 16-31: 32..63
    const uint4* src = (const uint4*)(Qb + (size_t)(qr + qrow) * DK_ + koff);
    union { uint4 q[2]; unsigned short s[32]; } u;
    u.q[0] = src[0]; u.q[1] = src[1];
#pragma unroll
    for (int e = 0; e < 32; ++e)
      qTw[(koff + e) * 16 + qrow] = u.s[e];
  }

  // ---- persistent B operands Q^T (dk 0..31 and 32..63) ----
  BFVec bq0, bq1;
  for (int s = 0; s < 8; ++s) {
    const int kk = (lane >> 4) * 8 + s;
    bq0.s[s]     = qTw[kk * 16 + (lane & 15)];
    bq0.s[8 + s] = qTw[(kk + 16) * 16 + (lane & 15)];
    bq1.s[s]     = qTw[(kk + 32) * 16 + (lane & 15)];
    bq1.s[8 + s] = qTw[(kk + 48) * 16 + (lane & 15)];
  }

  const float NEG_INF = -__builtin_inff();
  float m_state = NEG_INF;
  float l_state = 0.f;
  f32x8 o0 = {0.f,0.f,0.f,0.f,0.f,0.f,0.f,0.f};
  f32x8 o1 = o0, o2 = o0, o3 = o0;

  const int qg  = qr + (lane & 15);        // this lane's query index (N dim)
  const int nkb = blockIdx.x * 4 + 4;      // uniform over block (causal max)

  for (int kb = 0; kb < nkb; ++kb) {
    const int key0 = kb * 32;
    const unsigned buf = (unsigned)(kb & 1) * 4096u;

    // TDM data for this block is ready once TENSORcnt==0 on the issuing wave;
    // barrier publishes it to all waves in the workgroup.
    if (wave == 0) __builtin_amdgcn_s_wait_tensorcnt(0);
    __syncthreads();

    // prefetch next block into the other buffer while we compute
    if (wave == 0 && kb + 1 < nkb) {
      const unsigned nbuf = (unsigned)((kb + 1) & 1) * 4096u;
      tdm_load_2d(KT_OFF + nbuf, Kb + (size_t)(key0 + 32) * DK_, 32, DK_, S_);
      tdm_load_2d(VT_OFF + nbuf, Vb + (size_t)(key0 + 32) * DK_, 32, DK_, S_);
    }

    const unsigned short* kTw = (const unsigned short*)(smem + KT_OFF + buf);
    const unsigned short* vTw = (const unsigned short*)(smem + VT_OFF + buf);

    // ---- S^T tiles: A = K rows (LDS, contiguous), B = Q^T ----
    f32x8 sc[2];
#pragma unroll
    for (int t = 0; t < 2; ++t) {
      const unsigned short* kp =
          kTw + (size_t)(t * 16 + (lane & 15)) * DK_ + (lane >> 4) * 8;
      BFVec alo, ahi;
      alo.q[0] = *(const uint4*)(kp);
      alo.q[1] = *(const uint4*)(kp + 16);
      ahi.q[0] = *(const uint4*)(kp + 32);
      ahi.q[1] = *(const uint4*)(kp + 48);
      f32x8 c = {0.f,0.f,0.f,0.f,0.f,0.f,0.f,0.f};
      c = wmma_bf16(alo, bq0, c);
      c = wmma_bf16(ahi, bq1, c);
      sc[t] = c;
    }

    // ---- scale, causal mask, running max over keys ----
    float tmax = NEG_INF;
#pragma unroll
    for (int t = 0; t < 2; ++t)
#pragma unroll
      for (int r = 0; r < 8; ++r) {
        const int key = key0 + t * 16 + r + 8 * (lane >> 4);
        float sv = sc[t][r] * SCALE_;
        sv = (key <= qg) ? sv : NEG_INF;
        sc[t][r] = sv;
        tmax = fmaxf(tmax, sv);
      }
    tmax = fmaxf(tmax, __shfl_xor(tmax, 16, 32));
    const float m_new = fmaxf(m_state, tmax);
    const float corr  = __expf(m_state - m_new);
    float rsum = 0.f;
#pragma unroll
    for (int t = 0; t < 2; ++t)
#pragma unroll
      for (int r = 0; r < 8; ++r) {
        const float p = __expf(sc[t][r] - m_new);
        sc[t][r] = p;
        rsum += p;
      }
    rsum += __shfl_xor(rsum, 16, 32);
    l_state = l_state * corr + rsum;
    m_state = m_new;

    // ---- P^T (C layout) -> per-wave LDS as P[q][key] in bf16 ----
    {
      const int q = lane & 15;
#pragma unroll
      for (int t = 0; t < 2; ++t) {
        const int keybase = t * 16 + 8 * (lane >> 4);
        uint4 pk;
        pk.x = pack_bf16x2(sc[t][0], sc[t][1]);
        pk.y = pack_bf16x2(sc[t][2], sc[t][3]);
        pk.z = pack_bf16x2(sc[t][4], sc[t][5]);
        pk.w = pack_bf16x2(sc[t][6], sc[t][7]);
        *(uint4*)(pSw + q * 32 + keybase) = pk;
      }
    }

    // ---- rescale O accumulators (per-row corr via lane permute) ----
#pragma unroll
    for (int r = 0; r < 8; ++r) {
      const int srcl = (r + 8 * (lane >> 4)) | (lane & 16);
      const float cr = __shfl(corr, srcl, 32);
      o0[r] *= cr; o1[r] *= cr; o2[r] *= cr; o3[r] *= cr;
    }

    // ---- PV: A = P (LDS, A layout), B = V (LDS, contiguous rows) ----
    BFVec aP;
    {
      const int qa = lane & 15;
      const int ka = (lane >> 4) * 8;
      aP.q[0] = *(const uint4*)(pSw + qa * 32 + ka);
      aP.q[1] = *(const uint4*)(pSw + qa * 32 + ka + 16);
    }
    const unsigned short* vp = vTw + (size_t)((lane >> 4) * 8) * DK_ + (lane & 15);
    BFVec bv0, bv1, bv2, bv3;
    for (int s = 0; s < 8; ++s) {
      const unsigned short* v0 = vp + (size_t)s * DK_;
      const unsigned short* v1 = vp + (size_t)(16 + s) * DK_;
      bv0.s[s] = v0[0];  bv0.s[8 + s] = v1[0];
      bv1.s[s] = v0[16]; bv1.s[8 + s] = v1[16];
      bv2.s[s] = v0[32]; bv2.s[8 + s] = v1[32];
      bv3.s[s] = v0[48]; bv3.s[8 + s] = v1[48];
    }

    o0 = wmma_bf16(aP, bv0, o0);
    o1 = wmma_bf16(aP, bv1, o1);
    o2 = wmma_bf16(aP, bv2, o2);
    o3 = wmma_bf16(aP, bv3, o3);

    // All waves done reading this buffer before wave 0 can refill it on the
    // next iteration (the top-of-loop barrier provides that edge).
  }

  // ---- normalize and store O as bf16 [b, s, h*64+dk] ----
#pragma unroll
  for (int r = 0; r < 8; ++r) {
    const int srcl = (r + 8 * (lane >> 4)) | (lane & 16);
    const float lr  = __shfl(l_state, srcl, 32);
    const float inv = 1.0f / lr;
    const int srow  = qr + r + 8 * (lane >> 4);
    unsigned short* op =
        O + (size_t)(b * S_ + srow) * D_ + h * DK_ + (lane & 15);
    op[0]  = f32_to_bf16(o0[r] * inv);
    op[16] = f32_to_bf16(o1[r] * inv);
    op[32] = f32_to_bf16(o2[r] * inv);
    op[48] = f32_to_bf16(o3[r] * inv);
  }
}

// ---------------------------------------------------------------------------
// Output projection: O_bf16 @ Wo_bf16 + bo -> f32 out
// ---------------------------------------------------------------------------
__global__ void __launch_bounds__(256)
out_proj_gemm(const unsigned short* __restrict__ Obf,
              const unsigned short* __restrict__ wo,
              const float* __restrict__ bo,
              float* __restrict__ out) {
  const int lane = threadIdx.x & 31;
  const int wave = threadIdx.x >> 5;
  const int gt   = blockIdx.x * 8 + wave;
  const int TN = D_ / 16;
  const int m0 = (gt / TN) * 16;
  const int n0 = (gt % TN) * 16;

  f32x8 acc = gemm16x16(Obf, wo, m0, n0, lane);

  const int n = n0 + (lane & 15);
  const float bn = bo[n];
#pragma unroll
  for (int r = 0; r < 8; ++r) {
    const int m = m0 + r + 8 * (lane >> 4);
    out[(size_t)m * D_ + n] = acc[r] + bn;
  }
}

// ---------------------------------------------------------------------------
// Host entry. Workspace layout (bytes):
//  [0,8M)    x_bf16     [8M,10M)  Wq_bf16  [10M,12M) Wk_bf16
//  [12M,14M) Wv_bf16    [14M,16M) Wo_bf16
//  [16M,24M) Q_bf16     [24M,32M) K_bf16   [32M,40M) V_bf16  [40M,48M) O_bf16
// ---------------------------------------------------------------------------
extern "C" void kernel_launch(void* const* d_in, const int* in_sizes, int n_in,
                              void* d_out, int out_size, void* d_ws, size_t ws_size,
                              hipStream_t stream) {
  (void)in_sizes; (void)n_in; (void)out_size; (void)ws_size;
  const float* x  = (const float*)d_in[0];
  const int*   tp = (const int*)d_in[1];
  const float* Wq = (const float*)d_in[2];
  const float* bq = (const float*)d_in[3];
  const float* Wk = (const float*)d_in[4];
  const float* bk = (const float*)d_in[5];
  const float* Wv = (const float*)d_in[6];
  const float* bv = (const float*)d_in[7];
  const float* Wo = (const float*)d_in[8];
  const float* bo = (const float*)d_in[9];
  float* out = (float*)d_out;

  char* ws = (char*)d_ws;
  const size_t MB = (size_t)1 << 20;
  unsigned short* xbf = (unsigned short*)(ws + 0 * MB);
  unsigned short* wqb = (unsigned short*)(ws + 8 * MB);
  unsigned short* wkb = (unsigned short*)(ws + 10 * MB);
  unsigned short* wvb = (unsigned short*)(ws + 12 * MB);
  unsigned short* wob = (unsigned short*)(ws + 14 * MB);
  unsigned short* Qb  = (unsigned short*)(ws + 16 * MB);
  unsigned short* Kb  = (unsigned short*)(ws + 24 * MB);
  unsigned short* Vb  = (unsigned short*)(ws + 32 * MB);
  unsigned short* Ob  = (unsigned short*)(ws + 40 * MB);

  cvt_f32_bf16<<<(M_ * D_) / 1024, 256, 0, stream>>>(x,  xbf, M_ * D_);
  cvt_f32_bf16<<<(D_ * D_) / 1024, 256, 0, stream>>>(Wq, wqb, D_ * D_);
  cvt_f32_bf16<<<(D_ * D_) / 1024, 256, 0, stream>>>(Wk, wkb, D_ * D_);
  cvt_f32_bf16<<<(D_ * D_) / 1024, 256, 0, stream>>>(Wv, wvb, D_ * D_);
  cvt_f32_bf16<<<(D_ * D_) / 1024, 256, 0, stream>>>(Wo, wob, D_ * D_);

  // 3 matrices * 256 m-tiles * 64 n-tiles = 49152 wave tiles / 8 waves
  qkv_rope_gemm<<<(3 * 256 * 64) / 8, 256, 0, stream>>>(
      xbf, wqb, wkb, wvb, bq, bk, bv, tp, Qb, Kb, Vb);

  flash_attn<<<dim3(S_ / 128, H_, B_), 256, 0, stream>>>(Qb, Kb, Vb, Ob);

  out_proj_gemm<<<(256 * 64) / 8, 256, 0, stream>>>(Ob, wob, bo, out);
}
